// StateSpaceMixerModule_76295799046724
// MI455X (gfx1250) — compile-verified, matrix operand order checked by reading, not run on hardware
//
#include <hip/hip_runtime.h>
#include <hip/hip_bf16.h>

typedef unsigned short u16;
typedef __attribute__((ext_vector_type(16))) __bf16 v16bf;
typedef __attribute__((ext_vector_type(8)))  float  v8f;

// ---------------------------------------------------------------------------
// bf16 helpers (round-to-nearest-even)
// ---------------------------------------------------------------------------
static __device__ __forceinline__ u16 f2bf(float x) {
    unsigned int u = __float_as_uint(x);
    u += 0x7fffu + ((u >> 16) & 1u);
    return (u16)(u >> 16);
}

__global__ __launch_bounds__(256) void cvt_bf16_kernel(const float* __restrict__ src,
                                                       u16* __restrict__ dst, int n4) {
    int i = blockIdx.x * blockDim.x + threadIdx.x;
    if (i >= n4) return;
    float4 f = ((const float4*)src)[i];
    ushort4 h;
    h.x = f2bf(f.x); h.y = f2bf(f.y); h.z = f2bf(f.z); h.w = f2bf(f.w);
    ((ushort4*)dst)[i] = h;
}

// ---------------------------------------------------------------------------
// WMMA fragment: 16x32 bf16 tile slice per ISA 7.12.2 layout.
// lane&15 selects the row (A) / column (B); lane>>4 selects which K-half.
// elements 0-7  (VGPR0-3) <- K = k0 + kh*8      .. +7
// elements 8-15 (VGPR4-7) <- K = k0 + 16 + kh*8 .. +7
// ---------------------------------------------------------------------------
union Frag {
    v16bf v;
    uint4 q[2];
};

static __device__ __forceinline__ Frag load_frag(const u16* __restrict__ base,
                                                 int row, int K, int k0, int kh) {
    Frag f;
    const u16* p = base + (size_t)row * K + k0 + kh * 8;
    f.q[0] = *(const uint4*)(p);
    f.q[1] = *(const uint4*)(p + 16);
    return f;
}

static __device__ __forceinline__ v8f wmma_bf16(const Frag& a, const Frag& b, v8f c) {
    return __builtin_amdgcn_wmma_f32_16x16x32_bf16(
        /*neg_a=*/false, a.v, /*neg_b=*/false, b.v,
        /*c_mod=*/(short)0, c, /*reuse_a=*/false, /*reuse_b=*/false);
}

// ---------------------------------------------------------------------------
// One 64x64 output tile per wave: 4x4 WMMA tiles, K-loop in steps of 32.
// Per K-step: 4 A-frag loads + 4 B-frag loads feed 16 WMMAs
// (512 B of operand traffic per WMMA = 32 FLOP/byte).
// Live fragments kept to 5 (4 A + 1 B) to bound VGPR pressure.
// ---------------------------------------------------------------------------
static __device__ __forceinline__ void mma_block_64x64(
    const u16* __restrict__ Abase, const u16* __restrict__ Bbase,
    int m0, int n0, int K, int lr, int kh, v8f acc[4][4])
{
    for (int k0 = 0; k0 < K; k0 += 32) {
        Frag a[4];
        #pragma unroll
        for (int r = 0; r < 4; ++r)
            a[r] = load_frag(Abase, m0 + r * 16 + lr, K, k0, kh);
        #pragma unroll
        for (int c = 0; c < 4; ++c) {
            Frag b = load_frag(Bbase, n0 + c * 16 + lr, K, k0, kh);
            #pragma unroll
            for (int r = 0; r < 4; ++r)
                acc[r][c] = wmma_bf16(a[r], b, acc[r][c]);
        }
    }
}

// ---------------------------------------------------------------------------
// GEMM 1: Bu[M x 256] = Xb[M x 1024] * Bw[256 x 1024]^T + B_b
// 4 waves/block arranged 2x2 -> 128x128 block tile.
// ---------------------------------------------------------------------------
__global__ __launch_bounds__(128) void gemm_bu_kernel(
    const u16* __restrict__ Xb, const u16* __restrict__ Wb,
    const float* __restrict__ bias, float* __restrict__ out,
    int M, int N, int K)
{
    const int wave = threadIdx.x >> 5;
    const int lane = threadIdx.x & 31;
    const int m0   = blockIdx.x * 128 + (wave & 1) * 64;
    const int n0   = blockIdx.y * 128 + (wave >> 1) * 64;
    const int lr   = lane & 15;
    const int kh   = lane >> 4;

    v8f acc[4][4] = {};
    mma_block_64x64(Xb, Wb, m0, n0, K, lr, kh, acc);

    #pragma unroll
    for (int r = 0; r < 4; ++r)
        #pragma unroll
        for (int c = 0; c < 4; ++c) {
            const int col = n0 + c * 16 + lr;
            const float bv = bias[col];
            const int rowbase = m0 + r * 16 + kh * 8;
            #pragma unroll
            for (int i = 0; i < 8; ++i)
                out[(size_t)(rowbase + i) * N + col] = acc[r][c][i] + bv;
        }
}

// ---------------------------------------------------------------------------
// Scan: state = tanh(state * exp(A[n]) + Bu[b,t,n]); one lane per (b,n) chain.
// 4 blocks x 256 threads = the full 1024 independent recurrences.
// Latency-bound: v_tanh_f32 dependence chain; prefetch Bu 8 steps ahead.
// States written out as bf16 for the output GEMM.
// ---------------------------------------------------------------------------
__global__ __launch_bounds__(256) void scan_kernel(const float* __restrict__ Bu,
                                                   const float* __restrict__ A,
                                                   u16* __restrict__ Sb, int S)
{
    const int b = blockIdx.x;
    const int n = threadIdx.x;           // 0..255 = STATE index
    const float aexp = expf(A[n]);
    float state = 0.0f;

    const float* bu = Bu + (size_t)b * S * 256 + n;
    u16*         sp = Sb + (size_t)b * S * 256 + n;

    for (int t = 0; t < S; ++t) {
        __builtin_prefetch(bu + 256 * 8, 0, 0);     // global_prefetch_b8
        state = tanhf(state * aexp + bu[0]);
        sp[0] = f2bf(state);
        bu += 256;
        sp += 256;
    }
}

// ---------------------------------------------------------------------------
// GEMM 2 (fused epilogue): y[M x 1024] =
//     states_bf16[M x 256] * Cw[1024 x 256]^T
//   + Xb[M x 1024]        * Dw[1024 x 1024]^T
//   + C_b + D_b
// Two K-loops accumulate into the same 64x64 wave tile.
// ---------------------------------------------------------------------------
__global__ __launch_bounds__(128) void gemm_out_kernel(
    const u16* __restrict__ Sb,  const u16* __restrict__ Cwb,
    const u16* __restrict__ Xb,  const u16* __restrict__ Dwb,
    const float* __restrict__ Cb, const float* __restrict__ Db,
    float* __restrict__ out, int M)
{
    const int N = 1024;
    const int wave = threadIdx.x >> 5;
    const int lane = threadIdx.x & 31;
    const int m0   = blockIdx.x * 128 + (wave & 1) * 64;
    const int n0   = blockIdx.y * 128 + (wave >> 1) * 64;
    const int lr   = lane & 15;
    const int kh   = lane >> 4;

    v8f acc[4][4] = {};
    mma_block_64x64(Sb, Cwb, m0, n0, 256,  lr, kh, acc);   // states * C_w^T
    mma_block_64x64(Xb, Dwb, m0, n0, 1024, lr, kh, acc);   // x * D_w^T

    #pragma unroll
    for (int r = 0; r < 4; ++r)
        #pragma unroll
        for (int c = 0; c < 4; ++c) {
            const int col = n0 + c * 16 + lr;
            const float bv = Cb[col] + Db[col];
            const int rowbase = m0 + r * 16 + kh * 8;
            #pragma unroll
            for (int i = 0; i < 8; ++i)
                out[(size_t)(rowbase + i) * N + col] = acc[r][c][i] + bv;
        }
}

// ---------------------------------------------------------------------------
// Host entry
// ---------------------------------------------------------------------------
extern "C" void kernel_launch(void* const* d_in, const int* in_sizes, int n_in,
                              void* d_out, int out_size, void* d_ws, size_t ws_size,
                              hipStream_t stream) {
    (void)in_sizes; (void)n_in; (void)out_size; (void)ws_size;

    const float* x  = (const float*)d_in[0];
    const float* A  = (const float*)d_in[1];
    const float* Bw = (const float*)d_in[2];
    const float* Bb = (const float*)d_in[3];
    const float* Cw = (const float*)d_in[4];
    const float* Cb = (const float*)d_in[5];
    const float* Dw = (const float*)d_in[6];
    const float* Db = (const float*)d_in[7];
    float* y = (float*)d_out;

    const int    Bsz = 4, S = 4096, DIM = 1024, ST = 256;
    const size_t M   = (size_t)Bsz * S;   // 16384

    // Workspace layout (256B aligned slices), ~59 MB total
    char* ws = (char*)d_ws;
    auto alloc = [&](size_t bytes) {
        char* p = ws;
        ws += (bytes + 255) & ~(size_t)255;
        return p;
    };
    u16*   xb  = (u16*)alloc(M * DIM * sizeof(u16));           // 32 MB
    u16*   bwb = (u16*)alloc((size_t)ST * DIM * sizeof(u16));  // 512 KB
    u16*   cwb = (u16*)alloc((size_t)DIM * ST * sizeof(u16));  // 512 KB
    u16*   dwb = (u16*)alloc((size_t)DIM * DIM * sizeof(u16)); // 2 MB
    float* bu  = (float*)alloc(M * ST * sizeof(float));        // 16 MB
    u16*   sb  = (u16*)alloc(M * ST * sizeof(u16));            // 8 MB

    // 1) fp32 -> bf16 converts (x read once from HBM here)
    auto cvt = [&](const float* s, u16* d, size_t n) {
        int n4 = (int)(n / 4);
        cvt_bf16_kernel<<<dim3((n4 + 255) / 256), dim3(256), 0, stream>>>(s, d, n4);
    };
    cvt(x,  xb,  M * DIM);
    cvt(Bw, bwb, (size_t)ST * DIM);
    cvt(Cw, cwb, (size_t)DIM * ST);
    cvt(Dw, dwb, (size_t)DIM * DIM);

    // 2) Bu = x @ Bw^T + Bb   (WMMA bf16, 128x128 block tiles)
    gemm_bu_kernel<<<dim3((unsigned)(M / 128), ST / 128), dim3(128), 0, stream>>>(
        xb, bwb, Bb, bu, (int)M, ST, DIM);

    // 3) nonlinear recurrence (latency-bound chain)
    scan_kernel<<<dim3(Bsz), dim3(ST), 0, stream>>>(bu, A, sb, S);

    // 4) y = states @ Cw^T + x @ Dw^T + Cb + Db   (WMMA bf16, fused)
    gemm_out_kernel<<<dim3((unsigned)(M / 128), DIM / 128), dim3(128), 0, stream>>>(
        sb, cwb, xb, dwb, Cb, Db, y, (int)M);
}
